// AttentionLayer_9208409882840
// MI455X (gfx1250) — matrix operand order repaired and checked by reading, not compiled, not run on hardware
//
#include <hip/hip_runtime.h>

// ---------------------------------------------------------------------------
// Transformer layer (pre-norm attn w/ QK-norm + RoPE, SwiGLU FFN) for gfx1250.
// All GEMMs via v_wmma_f32_16x16x32_bf16 (wave32). ~172 GFLOP vs ~100MB of
// traffic -> matrix-core bound, so bf16 operands + f32 WMMA accumulation.
// GEMM: 128x128 block tile, 8 waves, each wave owns 32x64 (8 WMMA/K-step),
// double-buffered LDS fed by GLOBAL_LOAD_ASYNC_TO_LDS_B128 (ASYNCcnt) when
// the toolchain exposes it, register staging otherwise.
// ---------------------------------------------------------------------------

typedef __attribute__((ext_vector_type(16))) __bf16 v16bf;
typedef __attribute__((ext_vector_type(8)))  __bf16 v8bf;
typedef __attribute__((ext_vector_type(8)))  float  v8f;
typedef __attribute__((ext_vector_type(4)))  float  v4f;
typedef __attribute__((ext_vector_type(4)))  __bf16 v4bf;
typedef __attribute__((ext_vector_type(4)))  int    v4i;

// Address-space-qualified int32x4 pointers for the async-LDS builtin
// (clang's prototype: (AS1 v4i*, AS3 v4i*, imm offset, imm cpol)).
typedef __attribute__((address_space(1))) v4i gas_v4i;
typedef __attribute__((address_space(3))) v4i las_v4i;

constexpr int DIMc = 1024, HEADSc = 16, HDc = 64, FFNc = 4096, Bc = 2, Sc = 2048;
constexpr int TOKENS = Bc * Sc;               // 4096

#if defined(__has_builtin)
#if __has_builtin(__builtin_amdgcn_global_load_async_to_lds_b128)
#define HAVE_ASYNC_LDS 1
#endif
#endif
#ifndef HAVE_ASYNC_LDS
#define HAVE_ASYNC_LDS 0
#endif

// 16B global -> LDS copy: async (cache->LDS, ASYNCcnt) when available.
__device__ inline void copy16_g2l(const __bf16* g, __bf16* l) {
#if HAVE_ASYNC_LDS
  __builtin_amdgcn_global_load_async_to_lds_b128(
      (gas_v4i*)(void*)g, (las_v4i*)(void*)l, 0, 0);
#else
  *(v8bf*)l = *(const v8bf*)g;
#endif
}

__device__ inline void wait_async_copies() {
#if HAVE_ASYNC_LDS
#if __has_builtin(__builtin_amdgcn_s_wait_asynccnt)
  __builtin_amdgcn_s_wait_asynccnt(0);
#else
  asm volatile("s_wait_asynccnt 0" ::: "memory");
#endif
#endif
}

// Build a 16-element A/B fragment from two 16-byte LDS chunks.
__device__ inline v16bf load_frag(const __bf16* p0, const __bf16* p1) {
  v8bf a = *(const v8bf*)p0;
  v8bf b = *(const v8bf*)p1;
  return __builtin_shufflevector(a, b, 0,1,2,3,4,5,6,7,8,9,10,11,12,13,14,15);
}

__device__ inline v8f wmma_bf16(v16bf a, v16bf b, v8f c) {
  return __builtin_amdgcn_wmma_f32_16x16x32_bf16(false, a, false, b, (short)0, c,
                                                 false, false);
}

// 8-wave (256-thread) block reduction of two partial sums.
__device__ inline void block_reduce2(float& s, float& s2, float* red) {
  for (int off = 16; off > 0; off >>= 1) {
    s  += __shfl_xor(s,  off, 32);
    s2 += __shfl_xor(s2, off, 32);
  }
  const int wid = threadIdx.x >> 5;
  __syncthreads();                       // allow red[] reuse across calls
  if ((threadIdx.x & 31) == 0) { red[wid] = s; red[8 + wid] = s2; }
  __syncthreads();
  s = 0.f; s2 = 0.f;
  for (int i = 0; i < 8; ++i) { s += red[i]; s2 += red[8 + i]; }
}

// ---------------------------------------------------------------------------
// Elementwise: fp32 -> bf16 (vectorized x4; all sizes are multiples of 4)
// ---------------------------------------------------------------------------
__global__ void cvt_f32_bf16(const float* __restrict__ in, __bf16* __restrict__ out,
                             size_t n4) {
  for (size_t i = (size_t)blockIdx.x * blockDim.x + threadIdx.x; i < n4;
       i += (size_t)gridDim.x * blockDim.x) {
    v4f v = *(const v4f*)(in + 4 * i);
    v4bf o = { (__bf16)v.x, (__bf16)v.y, (__bf16)v.z, (__bf16)v.w };
    *(v4bf*)(out + 4 * i) = o;
  }
}

// ---------------------------------------------------------------------------
// Row LayerNorm, bf16 output (one block per row of length D)
// ---------------------------------------------------------------------------
__global__ __launch_bounds__(256)
void layernorm_bf16(const float* __restrict__ x, const float* __restrict__ g,
                    const float* __restrict__ b, __bf16* __restrict__ out, int D) {
  __shared__ float red[16];
  const size_t row = blockIdx.x;
  const float* xr = x + row * D;
  float s = 0.f, s2 = 0.f;
  for (int i = threadIdx.x; i < D; i += 256) { float v = xr[i]; s += v; s2 += v * v; }
  block_reduce2(s, s2, red);
  const float mean = s / D;
  const float inv  = rsqrtf(s2 / D - mean * mean + 1e-6f);
  for (int i = threadIdx.x; i < D; i += 256)
    out[row * D + i] = (__bf16)((xr[i] - mean) * inv * g[i] + b[i]);
}

// ---------------------------------------------------------------------------
// Tiled WMMA GEMM:  out[M,N] = A[M,K](bf16) * W[N,K]^T(bf16) + bias (+resid)
// Block tile 128x128, K-step 32, 8 waves in a 4x2 grid; each wave owns a
// 32x64 strip = 2x4 WMMA accumulator tiles (8 WMMA per K-step, 6 frag loads).
// Double-buffered LDS; tile k+1 copied (async) while tile k is computed.
// ---------------------------------------------------------------------------
template <int FUSE_RESID>
__global__ __launch_bounds__(256)
void gemm_bf16_wmma(const __bf16* __restrict__ A, const __bf16* __restrict__ W,
                    const float* __restrict__ bias, const float* __restrict__ resid,
                    float* __restrict__ out, int M, int N, int K) {
  __shared__ __bf16 As[2][128][32];
  __shared__ __bf16 Bs[2][128][32];
  const int tid = threadIdx.x, lane = tid & 31, wid = tid >> 5;
  const int mi = wid & 3, ni = wid >> 2;            // 4 x 2 wave grid
  const int lh = lane >> 4, l16 = lane & 15;        // lane half / lane%16
  const int tileM = blockIdx.y * 128, tileN = blockIdx.x * 128;
  const int srow = tid >> 1, sk = (tid & 1) * 16;   // staging: 32B/thread/matrix

  const __bf16* Aptr = &A[(size_t)(tileM + srow) * K + sk];
  const __bf16* Wptr = &W[(size_t)(tileN + srow) * K + sk];

  v8f acc[2][4] = {};                               // [m-subtile][n-subtile]

  // prologue: stage tile 0
  copy16_g2l(Aptr,     &As[0][srow][sk]);
  copy16_g2l(Aptr + 8, &As[0][srow][sk + 8]);
  copy16_g2l(Wptr,     &Bs[0][srow][sk]);
  copy16_g2l(Wptr + 8, &Bs[0][srow][sk + 8]);
  wait_async_copies();
  __syncthreads();

  int buf = 0;
  for (int k0 = 0; k0 < K; k0 += 32) {
    const bool has_next = (k0 + 32) < K;
    if (has_next) {                                  // stage tile k+1 -> buf^1
      copy16_g2l(Aptr + k0 + 32, &As[buf ^ 1][srow][sk]);
      copy16_g2l(Aptr + k0 + 40, &As[buf ^ 1][srow][sk + 8]);
      copy16_g2l(Wptr + k0 + 32, &Bs[buf ^ 1][srow][sk]);
      copy16_g2l(Wptr + k0 + 40, &Bs[buf ^ 1][srow][sk + 8]);
      if (k0 + 64 < K) {                             // warm L2 for tile k+2
        __builtin_prefetch(Aptr + k0 + 64, 0, 1);
        __builtin_prefetch(Wptr + k0 + 64, 0, 1);
      }
    }

    // fragments in contiguous k order, k = 16*lh + s for A and B (all 16B-aligned ds_read_b128)
    v16bf af[2], bfr[4];
    for (int m = 0; m < 2; ++m) {
      const int am = mi * 32 + m * 16 + l16;
      af[m] = load_frag(&As[buf][am][lh ? 16 : 0], &As[buf][am][lh ? 24 : 8]);
    }
    const int bk = lh ? 16 : 0;
    for (int n = 0; n < 4; ++n) {
      const int bn = ni * 64 + n * 16 + l16;
      bfr[n] = load_frag(&Bs[buf][bn][bk], &Bs[buf][bn][bk + 8]);
    }
    for (int m = 0; m < 2; ++m)
      for (int n = 0; n < 4; ++n)
        acc[m][n] = wmma_bf16(af[m], bfr[n], acc[m][n]);

    if (has_next) wait_async_copies();
    __syncthreads();
    buf ^= 1;
  }

  // epilogue (C layout: lane<16 -> rows 0..7, lane>=16 -> rows 8..15)
  for (int m = 0; m < 2; ++m) {
    const int rowB = tileM + mi * 32 + m * 16 + (lh ? 8 : 0);
    for (int n = 0; n < 4; ++n) {
      const int col = tileN + ni * 64 + n * 16 + l16;
      const float bv = bias[col];
      for (int i = 0; i < 8; ++i) {
        const size_t idx = (size_t)(rowB + i) * N + col;
        float v = acc[m][n][i] + bv;
        if (FUSE_RESID) v += resid[idx];
        out[idx] = v;
      }
    }
  }
}

// ---------------------------------------------------------------------------
// QK-norm (full-dim LN) + RoPE + head-major repack; V repack.  One block/token.
//   qkv: (B*S, 3*DIM) f32  ->  q/k/v: (B,H,S,64) bf16
// ---------------------------------------------------------------------------
__global__ __launch_bounds__(256)
void qkv_norm_rope(const float* __restrict__ qkv,
                   const float* __restrict__ qn_g, const float* __restrict__ qn_b,
                   const float* __restrict__ kn_g, const float* __restrict__ kn_b,
                   const float* __restrict__ rc, const float* __restrict__ rs,
                   __bf16* __restrict__ qo, __bf16* __restrict__ ko,
                   __bf16* __restrict__ vo) {
  __shared__ float tmp[DIMc];
  __shared__ float red[16];
  const int token = blockIdx.x;
  const int b = token / Sc, s = token % Sc;

  for (int pass = 0; pass < 2; ++pass) {
    const float* xr = qkv + (size_t)token * (3 * DIMc) + pass * DIMc;
    const float* gg = pass ? kn_g : qn_g;
    const float* bb = pass ? kn_b : qn_b;
    __bf16* dst = pass ? ko : qo;
    float sum = 0.f, sq = 0.f;
    for (int i = threadIdx.x; i < DIMc; i += 256) { float v = xr[i]; sum += v; sq += v * v; }
    block_reduce2(sum, sq, red);
    const float mean = sum / DIMc;
    const float inv  = rsqrtf(sq / DIMc - mean * mean + 1e-6f);
    for (int i = threadIdx.x; i < DIMc; i += 256)
      tmp[i] = (xr[i] - mean) * inv * gg[i] + bb[i];
    __syncthreads();
    for (int i = threadIdx.x; i < DIMc; i += 256) {
      const int h = i >> 6, t = i & 63;
      const float xv  = tmp[i];
      const float rot = (t < 32) ? -tmp[(h << 6) + t + 32] : tmp[(h << 6) + t - 32];
      const float o   = xv * rc[s * HDc + t] + rot * rs[s * HDc + t];
      dst[((size_t)(b * HEADSc + h) * Sc + s) * HDc + t] = (__bf16)o;
    }
    __syncthreads();
  }
  const float* vr = qkv + (size_t)token * (3 * DIMc) + 2 * DIMc;
  for (int i = threadIdx.x; i < DIMc; i += 256) {
    const int h = i >> 6, t = i & 63;
    vo[((size_t)(b * HEADSc + h) * Sc + s) * HDc + t] = (__bf16)vr[i];
  }
}

// ---------------------------------------------------------------------------
// Flash attention, one block per (b,h, 64-row q-tile); 8 waves.
// QK^T and P*V both through WMMA; online softmax with O kept in WMMA C regs.
// ---------------------------------------------------------------------------
__global__ __launch_bounds__(256)
void flash_attn(const __bf16* __restrict__ q, const __bf16* __restrict__ k,
                const __bf16* __restrict__ v, __bf16* __restrict__ outp) {
  __shared__ __bf16 Qs[64][64];
  __shared__ __bf16 Ks[64][64];
  __shared__ __bf16 VTs[64][64];     // VTs[hd][key] = V[key][hd]
  __shared__ float  Sf[64][64];
  __shared__ __bf16 Ps[64][64];
  __shared__ float  m_s[64], l_s[64], a_s[64];

  const int qt = blockIdx.x, bh = blockIdx.y;
  const int b = bh / HEADSc, h = bh % HEADSc;
  const int q0 = qt * 64;
  const int tid = threadIdx.x, lane = tid & 31, wid = tid >> 5;
  const int mi = wid & 3, ni = wid >> 2, lh = lane >> 4, l16 = lane & 15;
  const __bf16* qb = q + ((size_t)bh * Sc + q0) * HDc;
  const __bf16* kb = k + (size_t)bh * Sc * HDc;
  const __bf16* vb = v + (size_t)bh * Sc * HDc;

  { // stage Q once (async path when available)
    const int r = tid >> 2, c = (tid & 3) * 16;
    copy16_g2l(&qb[r * HDc + c],     &Qs[r][c]);
    copy16_g2l(&qb[r * HDc + c + 8], &Qs[r][c + 8]);
  }
  if (tid < 64) { m_s[tid] = -1e30f; l_s[tid] = 0.f; }
  v8f o0 = {}, o1 = {};
  wait_async_copies();
  __syncthreads();

  const float scale = 0.125f;        // 1/sqrt(64)

  for (int kt = 0; kt < Sc / 64; ++kt) {
    { // stage K row-major (async) + V transposed (manual reshape copy)
      const int r = tid >> 2, c = (tid & 3) * 16;
      copy16_g2l(&kb[(size_t)(kt * 64 + r) * HDc + c],     &Ks[r][c]);
      copy16_g2l(&kb[(size_t)(kt * 64 + r) * HDc + c + 8], &Ks[r][c + 8]);
      for (int j = 0; j < 16; ++j)
        VTs[c + j][r] = vb[(size_t)(kt * 64 + r) * HDc + c + j];
    }
    wait_async_copies();
    __syncthreads();

    // ---- S = scale * Q K^T (wave's 16x32 strip, K-dim = hd = 64 -> 2 steps)
    v8f s0 = {}, s1 = {};
    const int am = mi * 16 + l16;
    for (int ks = 0; ks < 2; ++ks) {
      const int k32 = ks * 32;
      v16bf af = load_frag(&Qs[am][k32 + (lh ? 16 : 0)], &Qs[am][k32 + (lh ? 24 : 8)]);
      const int bn = ni * 32 + l16;
      const int bk = k32 + (lh ? 16 : 0);
      v16bf bf0 = load_frag(&Ks[bn][bk],      &Ks[bn][bk + 8]);
      v16bf bf1 = load_frag(&Ks[bn + 16][bk], &Ks[bn + 16][bk + 8]);
      s0 = wmma_bf16(af, bf0, s0);
      s1 = wmma_bf16(af, bf1, s1);
    }
    {
      const int rB = mi * 16 + (lh ? 8 : 0), cB = ni * 32 + l16;
      for (int i = 0; i < 8; ++i) {
        Sf[rB + i][cB]      = s0[i] * scale;
        Sf[rB + i][cB + 16] = s1[i] * scale;
      }
    }
    __syncthreads();

    // ---- online softmax, phase 1: row max (4 threads per row)
    {
      const int r = tid >> 2, qq = tid & 3;
      float mx = -1e30f;
      for (int j = 0; j < 16; ++j) mx = fmaxf(mx, Sf[r][qq * 16 + j]);
      mx = fmaxf(mx, __shfl_xor(mx, 1, 32));
      mx = fmaxf(mx, __shfl_xor(mx, 2, 32));
      if (qq == 0) {
        const float mo = m_s[r], mn = fmaxf(mo, mx);
        m_s[r] = mn;
        a_s[r] = __expf(mo - mn);
      }
    }
    __syncthreads();

    // ---- phase 2: P = exp(S - m), row sums, l update
    {
      const int r = tid >> 2, qq = tid & 3;
      const float mn = m_s[r];
      float ps = 0.f;
      for (int j = 0; j < 16; ++j) {
        const float p = __expf(Sf[r][qq * 16 + j] - mn);
        Ps[r][qq * 16 + j] = (__bf16)p;
        ps += p;
      }
      ps += __shfl_xor(ps, 1, 32);
      ps += __shfl_xor(ps, 2, 32);
      if (qq == 0) l_s[r] = l_s[r] * a_s[r] + ps;
    }
    __syncthreads();

    // ---- rescale O accumulators and O += P * V  (K-dim = key = 64 -> 2 steps)
    {
      const int rB = mi * 16 + (lh ? 8 : 0);
      for (int i = 0; i < 8; ++i) { const float a = a_s[rB + i]; o0[i] *= a; o1[i] *= a; }
      for (int ks = 0; ks < 2; ++ks) {
        const int k32 = ks * 32;
        v16bf af = load_frag(&Ps[am][k32 + (lh ? 16 : 0)], &Ps[am][k32 + (lh ? 24 : 8)]);
        const int bn = ni * 32 + l16;          // hd column
        const int bk = k32 + (lh ? 16 : 0);
        v16bf bf0 = load_frag(&VTs[bn][bk],      &VTs[bn][bk + 8]);
        v16bf bf1 = load_frag(&VTs[bn + 16][bk], &VTs[bn + 16][bk + 8]);
        o0 = wmma_bf16(af, bf0, o0);
        o1 = wmma_bf16(af, bf1, o1);
      }
    }
    __syncthreads();
  }

  { // epilogue: O / l -> attnout (b, s, h*64+hd) bf16
    const int rB = mi * 16 + (lh ? 8 : 0), cB = ni * 32 + l16;
    for (int i = 0; i < 8; ++i) {
      const float invl = 1.f / l_s[rB + i];
      const size_t base = ((size_t)b * Sc + q0 + rB + i) * DIMc + h * HDc;
      outp[base + cB]      = (__bf16)(o0[i] * invl);
      outp[base + cB + 16] = (__bf16)(o1[i] * invl);
    }
  }
}

// ---------------------------------------------------------------------------
// SwiGLU gate: out = silu(g1) * g3   (bf16 out, feeds w2 GEMM; vectorized x4)
// ---------------------------------------------------------------------------
__global__ void swiglu_gate(const float* __restrict__ g1, const float* __restrict__ g3,
                            __bf16* __restrict__ out, size_t n4) {
  for (size_t i = (size_t)blockIdx.x * blockDim.x + threadIdx.x; i < n4;
       i += (size_t)gridDim.x * blockDim.x) {
    v4f a = *(const v4f*)(g1 + 4 * i);
    v4f c = *(const v4f*)(g3 + 4 * i);
    v4bf o;
    for (int j = 0; j < 4; ++j) {
      const float sig = 1.f / (1.f + __expf(-a[j]));
      o[j] = (__bf16)(a[j] * sig * c[j]);
    }
    *(v4bf*)(out + 4 * i) = o;
  }
}

// ---------------------------------------------------------------------------
// Launcher
// ---------------------------------------------------------------------------
extern "C" void kernel_launch(void* const* d_in, const int* in_sizes, int n_in,
                              void* d_out, int out_size, void* d_ws, size_t ws_size,
                              hipStream_t stream) {
  const float* x     = (const float*)d_in[0];
  const float* rc    = (const float*)d_in[1];
  const float* rs    = (const float*)d_in[2];
  const float* w_qkv = (const float*)d_in[3];
  const float* b_qkv = (const float*)d_in[4];
  const float* w_out = (const float*)d_in[5];
  const float* b_out = (const float*)d_in[6];
  const float* qn_g  = (const float*)d_in[7];
  const float* qn_b  = (const float*)d_in[8];
  const float* kn_g  = (const float*)d_in[9];
  const float* kn_b  = (const float*)d_in[10];
  const float* ln1_g = (const float*)d_in[11];
  const float* ln1_b = (const float*)d_in[12];
  const float* ln2_g = (const float*)d_in[13];
  const float* ln2_b = (const float*)d_in[14];
  const float* w1    = (const float*)d_in[15];
  const float* b1    = (const float*)d_in[16];
  const float* w2    = (const float*)d_in[17];
  const float* b2    = (const float*)d_in[18];
  const float* w3    = (const float*)d_in[19];
  const float* b3    = (const float*)d_in[20];
  float* out = (float*)d_out;

  // ---- bump allocator over workspace
  char* ws = (char*)d_ws;
  size_t off = 0;
  auto alloc = [&](size_t bytes) -> char* {
    char* p = ws + off;
    off += (bytes + 255) & ~(size_t)255;
    return p;
  };
  __bf16* wqkv_bf = (__bf16*)alloc((size_t)3 * DIMc * DIMc * 2);
  __bf16* wout_bf = (__bf16*)alloc((size_t)DIMc * DIMc * 2);
  __bf16* w1_bf   = (__bf16*)alloc((size_t)FFNc * DIMc * 2);
  __bf16* w2_bf   = (__bf16*)alloc((size_t)DIMc * FFNc * 2);
  __bf16* w3_bf   = (__bf16*)alloc((size_t)FFNc * DIMc * 2);
  __bf16* h_bf    = (__bf16*)alloc((size_t)TOKENS * DIMc * 2);
  float*  qkv     = (float*) alloc((size_t)TOKENS * 3 * DIMc * 4);
  __bf16* q_bf    = (__bf16*)alloc((size_t)TOKENS * DIMc * 2);
  __bf16* k_bf    = (__bf16*)alloc((size_t)TOKENS * DIMc * 2);
  __bf16* v_bf    = (__bf16*)alloc((size_t)TOKENS * DIMc * 2);
  __bf16* attn_bf = (__bf16*)alloc((size_t)TOKENS * DIMc * 2);
  float*  g1      = (float*) alloc((size_t)TOKENS * FFNc * 4);
  float*  g3      = (float*) alloc((size_t)TOKENS * FFNc * 4);
  __bf16* gg_bf   = (__bf16*)alloc((size_t)TOKENS * FFNc * 2);

  // ---- 1. weights -> bf16
  cvt_f32_bf16<<<1024, 256, 0, stream>>>(w_qkv, wqkv_bf, (size_t)3 * DIMc * DIMc / 4);
  cvt_f32_bf16<<<1024, 256, 0, stream>>>(w_out, wout_bf, (size_t)DIMc * DIMc / 4);
  cvt_f32_bf16<<<1024, 256, 0, stream>>>(w1, w1_bf, (size_t)FFNc * DIMc / 4);
  cvt_f32_bf16<<<1024, 256, 0, stream>>>(w2, w2_bf, (size_t)DIMc * FFNc / 4);
  cvt_f32_bf16<<<1024, 256, 0, stream>>>(w3, w3_bf, (size_t)FFNc * DIMc / 4);

  // ---- 2. pre-attn LN
  layernorm_bf16<<<TOKENS, 256, 0, stream>>>(x, ln1_g, ln1_b, h_bf, DIMc);

  // ---- 3. QKV projection (4096x3072x1024)
  gemm_bf16_wmma<0><<<dim3(3 * DIMc / 128, TOKENS / 128), 256, 0, stream>>>(
      h_bf, wqkv_bf, b_qkv, nullptr, qkv, TOKENS, 3 * DIMc, DIMc);

  // ---- 4. QK-norm + RoPE + head-major repack
  qkv_norm_rope<<<TOKENS, 256, 0, stream>>>(qkv, qn_g, qn_b, kn_g, kn_b, rc, rs,
                                            q_bf, k_bf, v_bf);

  // ---- 5. attention
  flash_attn<<<dim3(Sc / 64, Bc * HEADSc), 256, 0, stream>>>(q_bf, k_bf, v_bf, attn_bf);

  // ---- 6. out projection + residual -> d_out (x after attention)
  gemm_bf16_wmma<1><<<dim3(DIMc / 128, TOKENS / 128), 256, 0, stream>>>(
      attn_bf, wout_bf, b_out, x, out, TOKENS, DIMc, DIMc);

  // ---- 7. pre-FFN LN (reuse h_bf)
  layernorm_bf16<<<TOKENS, 256, 0, stream>>>(out, ln2_g, ln2_b, h_bf, DIMc);

  // ---- 8/9. gate and up projections (4096x4096x1024)
  gemm_bf16_wmma<0><<<dim3(FFNc / 128, TOKENS / 128), 256, 0, stream>>>(
      h_bf, w1_bf, b1, nullptr, g1, TOKENS, FFNc, DIMc);
  gemm_bf16_wmma<0><<<dim3(FFNc / 128, TOKENS / 128), 256, 0, stream>>>(
      h_bf, w3_bf, b3, nullptr, g3, TOKENS, FFNc, DIMc);

  // ---- 10. SwiGLU gate
  swiglu_gate<<<4096, 256, 0, stream>>>(g1, g3, gg_bf, (size_t)TOKENS * FFNc / 4);

  // ---- 11. down projection + residual (reads/writes d_out elementwise 1:1)
  gemm_bf16_wmma<1><<<dim3(DIMc / 128, TOKENS / 128), 256, 0, stream>>>(
      gg_bf, w2_bf, b2, out, out, TOKENS, DIMc, FFNc);
}
